// Lightning_35433480192096
// MI455X (gfx1250) — matrix operand order repaired and checked by reading, not compiled, not run on hardware
//
#include <hip/hip_runtime.h>
#include <hip/hip_bf16.h>
#include <math.h>

typedef _Float16 f16;
typedef __attribute__((ext_vector_type(16))) _Float16 v16h;
typedef __attribute__((ext_vector_type(8)))  _Float16 h8;
typedef __attribute__((ext_vector_type(8)))  float    v8f;

#define D_DIM 512
#define F_DIMc 1024
#define T_DIM 1024
#define B_DIM 32
#define NROWS 1056   // 1024 rows of batch 31 + 32 last-timestep rows

// ---------------- weight convert + transpose: Wt[n*K+k] = (f16)W[k*N+n] ----
__global__ __launch_bounds__(256)
void convT_kernel(const float* __restrict__ W, f16* __restrict__ Wt, int K, int N) {
  long i = (long)blockIdx.x * blockDim.x + threadIdx.x;
  long total = (long)K * N;
  if (i >= total) return;
  int k = (int)(i / N), n = (int)(i % N);
  Wt[(long)n * K + k] = (f16)W[i];
}

// ---------------- gather the 1056 needed feature rows, f32 -> f16 ----------
__global__ __launch_bounds__(256)
void gather_kernel(const float* __restrict__ feat, f16* __restrict__ Xh) {
  int row = blockIdx.x;
  int b, t;
  if (row < T_DIM) { b = 31; t = row; } else { b = row - T_DIM; t = T_DIM - 1; }
  const float* src = feat + ((long)b * T_DIM + t) * F_DIMc;
  f16* dst = Xh + (long)row * F_DIMc;
  for (int c = threadIdx.x; c < F_DIMc; c += blockDim.x) dst[c] = (f16)src[c];
}

// ---------------- generic WMMA GEMM: C = act(A(MxK,f16) * Bt^T(NxK,f16)+b) -
// One 16x16 output tile per wave; K-loop unrolled x2 (all K are mult of 64).
// A frag: lane L holds row m=L&15, halves e=0..7 -> k+8g+e, e=8..15 -> k+16+8g+(e-8)
// B frag: lane L holds col n=L&15, halves e=0..7 -> k+8g+e, e=8..15 -> k+16+8g+(e-8)
__global__ __launch_bounds__(128)
void wmma_gemm_kernel(const f16* __restrict__ A, const f16* __restrict__ Bt,
                      const float* __restrict__ bias,
                      float* __restrict__ Cf, f16* __restrict__ Ch,
                      int M, int N, int K, int relu)
{
  const int lane  = threadIdx.x & 31;
  const int wave  = threadIdx.x >> 5;
  const int tilesN = N >> 4;
  const int tile  = blockIdx.x * (blockDim.x >> 5) + wave;
  const int tM = tile / tilesN;
  const int tN = tile - tM * tilesN;
  if (tM * 16 >= M) return;

  const int r = lane & 15;
  const int g = lane >> 4;

  const f16* pa = A  + (long)(tM * 16 + r) * K + 8 * g;
  const f16* pb = Bt + (long)(tN * 16 + r) * K + 8 * g;

  v8f acc = {};
  for (int k = 0; k < K; k += 64) {
    union { v16h v; h8 h[2]; } fa0, fb0, fa1, fb1;
    fa0.h[0] = *(const h8*)(pa + k);
    fa0.h[1] = *(const h8*)(pa + k + 16);
    fb0.h[0] = *(const h8*)(pb + k);
    fb0.h[1] = *(const h8*)(pb + k + 16);
    fa1.h[0] = *(const h8*)(pa + k + 32);
    fa1.h[1] = *(const h8*)(pa + k + 48);
    fb1.h[0] = *(const h8*)(pb + k + 32);
    fb1.h[1] = *(const h8*)(pb + k + 48);
    acc = __builtin_amdgcn_wmma_f32_16x16x32_f16(false, fa0.v, false, fb0.v,
                                                 (short)0, acc, false, false);
    acc = __builtin_amdgcn_wmma_f32_16x16x32_f16(false, fa1.v, false, fb1.v,
                                                 (short)0, acc, false, false);
  }

  const int n = tN * 16 + r;
  const float bv = bias ? bias[n] : 0.0f;
#pragma unroll
  for (int j = 0; j < 8; ++j) {
    const int m = tM * 16 + 8 * g + j;
    float v = acc[j] + bv;
    if (relu) v = fmaxf(v, 0.0f);
    if (Cf) Cf[(long)m * N + n] = v;
    if (Ch) Ch[(long)m * N + n] = (f16)v;
  }
}

// ---------------- add sinusoidal positional encoding, emit f32 + f16 -------
__global__ __launch_bounds__(256)
void posenc_kernel(const float* __restrict__ H2, float* __restrict__ Yf,
                   f16* __restrict__ Yh)
{
  int row = blockIdx.x;
  int t = (row < T_DIM) ? row : (T_DIM - 1);
  for (int c = threadIdx.x; c < D_DIM; c += blockDim.x) {
    int i = c & ~1;  // arange(0,d,2) value used for this channel
    float dv  = expf((float)i * (-9.210340371976184f / (float)D_DIM));
    float ang = (float)t * dv;
    float pe  = (c & 1) ? cosf(ang) : sinf(ang);
    long idx = (long)row * D_DIM + c;
    float v = H2[idx] + pe;
    Yf[idx] = v;
    Yh[idx] = (f16)v;
  }
}

// ---------------- differential dual-softmax attention, 32 query rows -------
// block b handles query row 32*b+31 of batch 31; K/V from all 1024 rows.
__global__ __launch_bounds__(256)
void attn_kernel(const float* __restrict__ QKV, const float* __restrict__ lamp,
                 float* __restrict__ Of, f16* __restrict__ Oh)
{
  __shared__ float q[2 * 256];
  __shared__ float s1[T_DIM];
  __shared__ float s2[T_DIM];
  __shared__ float red[256];

  const int tid = threadIdx.x;
  const int b = blockIdx.x;
  const int qrow = 32 * b + 31;
  const float lam = lamp[0];

  const float* qp = QKV + (long)qrow * 1536;
  q[tid]       = qp[tid];
  q[tid + 256] = qp[tid + 256];
  __syncthreads();

  const float scale = 0.08838834764831845f;  // 1/sqrt(512/4)
  for (int s = tid; s < T_DIM; s += 256) {
    const float4* kp = (const float4*)(QKV + (long)s * 1536 + 512);
    float d1 = 0.f, d2 = 0.f;
#pragma unroll 4
    for (int c = 0; c < 64; ++c) {
      float4 kv = kp[c];
      d1 = fmaf(q[4*c+0], kv.x, d1); d1 = fmaf(q[4*c+1], kv.y, d1);
      d1 = fmaf(q[4*c+2], kv.z, d1); d1 = fmaf(q[4*c+3], kv.w, d1);
      float4 kw = kp[64 + c];
      d2 = fmaf(q[256+4*c+0], kw.x, d2); d2 = fmaf(q[256+4*c+1], kw.y, d2);
      d2 = fmaf(q[256+4*c+2], kw.z, d2); d2 = fmaf(q[256+4*c+3], kw.w, d2);
    }
    s1[s] = d1 * scale;
    s2[s] = d2 * scale;
  }
  __syncthreads();

  // softmax over s1
  float m = -1e30f;
  for (int s = tid; s < T_DIM; s += 256) m = fmaxf(m, s1[s]);
  red[tid] = m; __syncthreads();
  for (int st = 128; st; st >>= 1) { if (tid < st) red[tid] = fmaxf(red[tid], red[tid+st]); __syncthreads(); }
  const float m1 = red[0]; __syncthreads();
  float sum = 0.f;
  for (int s = tid; s < T_DIM; s += 256) { float e = __expf(s1[s] - m1); s1[s] = e; sum += e; }
  red[tid] = sum; __syncthreads();
  for (int st = 128; st; st >>= 1) { if (tid < st) red[tid] += red[tid+st]; __syncthreads(); }
  const float inv1 = 1.f / red[0]; __syncthreads();

  // softmax over s2
  m = -1e30f;
  for (int s = tid; s < T_DIM; s += 256) m = fmaxf(m, s2[s]);
  red[tid] = m; __syncthreads();
  for (int st = 128; st; st >>= 1) { if (tid < st) red[tid] = fmaxf(red[tid], red[tid+st]); __syncthreads(); }
  const float m2 = red[0]; __syncthreads();
  sum = 0.f;
  for (int s = tid; s < T_DIM; s += 256) { float e = __expf(s2[s] - m2); s2[s] = e; sum += e; }
  red[tid] = sum; __syncthreads();
  for (int st = 128; st; st >>= 1) { if (tid < st) red[tid] += red[tid+st]; __syncthreads(); }
  const float inv2 = 1.f / red[0]; __syncthreads();

  for (int s = tid; s < T_DIM; s += 256) s1[s] = s1[s] * inv1 - lam * (s2[s] * inv2);
  __syncthreads();

  // O[b,:] = sum_s diff[s] * V[s,:]
  float a0 = 0.f, a1 = 0.f;
  for (int s = 0; s < T_DIM; ++s) {
    const float w = s1[s];
    const float* vp = QKV + (long)s * 1536 + 1024;
    a0 = fmaf(w, vp[tid], a0);
    a1 = fmaf(w, vp[tid + 256], a1);
  }
  Of[(long)b * 512 + tid]       = a0;
  Of[(long)b * 512 + tid + 256] = a1;
  Oh[(long)b * 512 + tid]       = (f16)a0;
  Oh[(long)b * 512 + tid + 256] = (f16)a1;
}

// ---------------- residual add + LayerNorm over D=512 ---------------------
__global__ __launch_bounds__(256)
void residual_ln_kernel(const float* __restrict__ X, const float* __restrict__ R,
                        const float* __restrict__ gam, const float* __restrict__ bet,
                        float* __restrict__ Yf, f16* __restrict__ Yh)
{
  __shared__ float red[256];
  const int row = blockIdx.x;
  const int t = threadIdx.x;
  const long base = (long)row * D_DIM;
  float v0 = X[base + t]       + R[base + t];
  float v1 = X[base + t + 256] + R[base + t + 256];
  red[t] = v0 + v1; __syncthreads();
  for (int st = 128; st; st >>= 1) { if (t < st) red[t] += red[t+st]; __syncthreads(); }
  const float mu = red[0] / (float)D_DIM; __syncthreads();
  const float d0 = v0 - mu, d1 = v1 - mu;
  red[t] = d0 * d0 + d1 * d1; __syncthreads();
  for (int st = 128; st; st >>= 1) { if (t < st) red[t] += red[t+st]; __syncthreads(); }
  const float rs = rsqrtf(red[0] / (float)D_DIM + 1e-5f);
  const float o0 = d0 * rs * gam[t]       + bet[t];
  const float o1 = d1 * rs * gam[t + 256] + bet[t + 256];
  if (Yf) { Yf[base + t] = o0; Yf[base + t + 256] = o1; }
  Yh[base + t]       = (f16)o0;
  Yh[base + t + 256] = (f16)o1;
}

// ======================= host side ========================================
extern "C" void kernel_launch(void* const* d_in, const int* in_sizes, int n_in,
                              void* d_out, int out_size, void* d_ws, size_t ws_size,
                              hipStream_t stream)
{
  (void)in_sizes; (void)n_in; (void)out_size; (void)ws_size;
  const float* features = (const float*)d_in[0];
  const float* enc_w1 = (const float*)d_in[1];  const float* enc_b1 = (const float*)d_in[2];
  const float* enc_w2 = (const float*)d_in[3];  const float* enc_b2 = (const float*)d_in[4];
  const float* qkv_w  = (const float*)d_in[5];  const float* qkv_b  = (const float*)d_in[6];
  const float* lam    = (const float*)d_in[7];
  const float* out_w  = (const float*)d_in[8];  const float* out_b  = (const float*)d_in[9];
  const float* ln1_g  = (const float*)d_in[10]; const float* ln1_b  = (const float*)d_in[11];
  const float* ffn_w1 = (const float*)d_in[12]; const float* ffn_b1 = (const float*)d_in[13];
  const float* ffn_w2 = (const float*)d_in[14]; const float* ffn_b2 = (const float*)d_in[15];
  const float* ln2_g  = (const float*)d_in[16]; const float* ln2_b  = (const float*)d_in[17];
  const float* act_w1 = (const float*)d_in[18]; const float* act_b1 = (const float*)d_in[19];
  const float* act_w2 = (const float*)d_in[20]; const float* act_b2 = (const float*)d_in[21];
  const float* act_w3 = (const float*)d_in[22]; const float* act_b3 = (const float*)d_in[23];
  const float* cr_w1  = (const float*)d_in[24]; const float* cr_b1  = (const float*)d_in[25];
  const float* cr_w2  = (const float*)d_in[26]; const float* cr_b2  = (const float*)d_in[27];
  const float* cr_w3  = (const float*)d_in[28]; const float* cr_b3  = (const float*)d_in[29];

  char* base = (char*)d_ws;
  size_t off = 0;
  auto alloc = [&](size_t bytes) -> void* {
    void* p = base + off;
    off += (bytes + 255) & ~(size_t)255;
    return p;
  };

  // f16 transposed weights
  f16* wt_enc1 = (f16*)alloc((size_t)1024*512*2);
  f16* wt_enc2 = (f16*)alloc((size_t)512*512*2);
  f16* wt_qkv  = (f16*)alloc((size_t)512*1536*2);
  f16* wt_out  = (f16*)alloc((size_t)512*512*2);
  f16* wt_f1   = (f16*)alloc((size_t)512*1024*2);
  f16* wt_f2   = (f16*)alloc((size_t)1024*512*2);
  f16* wt_a1   = (f16*)alloc((size_t)512*1024*2);
  f16* wt_a2   = (f16*)alloc((size_t)1024*2048*2);
  f16* wt_a3   = (f16*)alloc((size_t)2048*256*2);
  f16* wt_c1   = (f16*)alloc((size_t)512*1024*2);
  f16* wt_c2   = (f16*)alloc((size_t)1024*2048*2);
  f16* wt_c3   = (f16*)alloc((size_t)2048*256*2);
  // activations
  f16*   Xh   = (f16*)  alloc((size_t)NROWS*F_DIMc*2);
  f16*   H1h  = (f16*)  alloc((size_t)NROWS*D_DIM*2);
  float* H2f  = (float*)alloc((size_t)NROWS*D_DIM*4);
  float* Yf   = (float*)alloc((size_t)NROWS*D_DIM*4);
  f16*   Yh   = (f16*)  alloc((size_t)NROWS*D_DIM*2);
  float* QKVf = (float*)alloc((size_t)1024*1536*4);
  float* Of   = (float*)alloc((size_t)32*512*4);
  f16*   Oh   = (f16*)  alloc((size_t)32*512*2);
  float* Pf   = (float*)alloc((size_t)32*512*4);
  float* y1f  = (float*)alloc((size_t)32*512*4);
  f16*   y1h  = (f16*)  alloc((size_t)32*512*2);
  f16*   F1h  = (f16*)  alloc((size_t)32*1024*2);
  float* Zf   = (float*)alloc((size_t)32*512*4);
  f16*   y2h  = (f16*)  alloc((size_t)32*512*2);
  f16*   A1h  = (f16*)  alloc((size_t)32*1024*2);
  f16*   A2h  = (f16*)  alloc((size_t)32*2048*2);
  f16*   C1h  = (f16*)  alloc((size_t)32*1024*2);
  f16*   C2h  = (f16*)  alloc((size_t)32*2048*2);

  auto convT = [&](const float* W, f16* Wt, int K, int N) {
    long total = (long)K * N;
    int blocks = (int)((total + 255) / 256);
    convT_kernel<<<blocks, 256, 0, stream>>>(W, Wt, K, N);
  };
  auto gemm = [&](const f16* A, const f16* Bt, const float* bias,
                  float* Cf, f16* Ch, int M, int N, int K, int relu) {
    int tiles = (M / 16) * (N / 16);
    int blocks = (tiles + 3) / 4;        // 4 waves per block, 1 tile per wave
    wmma_gemm_kernel<<<blocks, 128, 0, stream>>>(A, Bt, bias, Cf, Ch, M, N, K, relu);
  };

  convT(enc_w1, wt_enc1, 1024, 512);
  convT(enc_w2, wt_enc2, 512, 512);
  convT(qkv_w,  wt_qkv,  512, 1536);
  convT(out_w,  wt_out,  512, 512);
  convT(ffn_w1, wt_f1,   512, 1024);
  convT(ffn_w2, wt_f2,   1024, 512);
  convT(act_w1, wt_a1,   512, 1024);
  convT(act_w2, wt_a2,   1024, 2048);
  convT(act_w3, wt_a3,   2048, 256);
  convT(cr_w1,  wt_c1,   512, 1024);
  convT(cr_w2,  wt_c2,   1024, 2048);
  convT(cr_w3,  wt_c3,   2048, 256);

  gather_kernel<<<NROWS, 256, 0, stream>>>(features, Xh);

  // encoder
  gemm(Xh,  wt_enc1, enc_b1, nullptr, H1h, NROWS, 512, 1024, 1);
  gemm(H1h, wt_enc2, enc_b2, H2f, nullptr, NROWS, 512, 512, 1);
  posenc_kernel<<<NROWS, 256, 0, stream>>>(H2f, Yf, Yh);

  // QKV on the 1024 batch-31 rows (first 1024 rows of Yh)
  gemm(Yh, wt_qkv, qkv_b, QKVf, nullptr, 1024, 1536, 512, 0);

  // differential dual-softmax attention -> 32 output rows
  attn_kernel<<<32, 256, 0, stream>>>(QKVf, lam, Of, Oh);

  // output projection + residual LN1
  gemm(Oh, wt_out, out_b, Pf, nullptr, 32, 512, 512, 0);
  residual_ln_kernel<<<32, 256, 0, stream>>>(Yf + (size_t)1024 * D_DIM, Pf,
                                             ln1_g, ln1_b, y1f, y1h);
  // FFN + residual LN2
  gemm(y1h, wt_f1, ffn_b1, nullptr, F1h, 32, 1024, 512, 1);
  gemm(F1h, wt_f2, ffn_b2, Zf, nullptr, 32, 512, 1024, 0);
  residual_ln_kernel<<<32, 256, 0, stream>>>(y1f, Zf, ln2_g, ln2_b, nullptr, y2h);

  // actor head -> d_out[0 .. 8192)
  gemm(y2h, wt_a1, act_b1, nullptr, A1h, 32, 1024, 512, 1);
  gemm(A1h, wt_a2, act_b2, nullptr, A2h, 32, 2048, 1024, 1);
  gemm(A2h, wt_a3, act_b3, (float*)d_out, nullptr, 32, 256, 2048, 0);
  // critic head -> d_out[8192 .. 16384)
  gemm(y2h, wt_c1, cr_b1, nullptr, C1h, 32, 1024, 512, 1);
  gemm(C1h, wt_c2, cr_b2, nullptr, C2h, 32, 2048, 1024, 1);
  gemm(C2h, wt_c3, cr_b3, (float*)d_out + 32 * 256, nullptr, 32, 256, 2048, 0);
}
